// VRNN_GMM_73031623901797
// MI455X (gfx1250) — matrix operand order repaired and checked by reading, not compiled, not run on hardware
//
#include <hip/hip_runtime.h>
#include <math.h>

typedef unsigned short u16;
typedef __attribute__((ext_vector_type(16))) __bf16 v16bf;
typedef __attribute__((ext_vector_type(8)))  float  v8f;

constexpr int BD = 256;   // batch
constexpr int TD = 256;   // time steps
constexpr int HD = 512;   // hidden
constexpr int ZD = 64;    // latent

__device__ __forceinline__ u16 f2bf(float f) {
    union { float f; unsigned u; } v; v.f = f;
    unsigned r = v.u + 0x7fffu + ((v.u >> 16) & 1u);   // round-to-nearest-even
    return (u16)(r >> 16);
}

__device__ __forceinline__ void wait_async0() {
#if __has_builtin(__builtin_amdgcn_s_wait_asynccnt)
    __builtin_amdgcn_s_wait_asynccnt(0);
#else
    asm volatile("s_wait_asynccnt 0x0" ::: "memory");
#endif
}

// ---------------------------------------------------------------------------
// bf16 WMMA GEMM: out[M=256,N] = act( A[M,K] @ W[N,K]^T + bias )
// Block tile 128x64, 4 waves, each wave owns a 32x64 strip (2x4 accumulators).
// Double-buffered LDS filled by GLOBAL_LOAD_ASYNC_TO_LDS_B128 (ASYNCcnt):
// no VGPR staging, async DMA of slice k+1 overlaps the 8 WMMAs of slice k.
// A may be split across two buffers at column splitK (concat without copies);
// splitK is a multiple of GBK and both halves share lda, so the buffer select
// is a uniform scalar per K-slice.
// flags: bit0 = relu, bit1 = fp32 output (else bf16)
// ---------------------------------------------------------------------------
constexpr int GBM = 128, GBN = 64, GBK = 32;
constexpr int LSTR = 40;   // LDS row stride (elems): 80B rows, 16B aligned, conflict-free b128

__global__ __launch_bounds__(128, 1)
void gemm_bf16_wmma(const u16* __restrict__ A0,
                    const u16* __restrict__ A1, int lda, int splitK,
                    const u16* __restrict__ W,  int ldw,
                    const float* __restrict__ bias,
                    void* __restrict__ outp, int ldo,
                    int N, int K, int flags)
{
    __shared__ u16 sA[2][GBM * LSTR];
    __shared__ u16 sW[2][GBN * LSTR];

    const int tid  = threadIdx.x;
    const int lane = tid & 31;
    const int wave = tid >> 5;           // wave strip: rows [wave*32, wave*32+32)
    const int mt0  = blockIdx.y * GBM;
    const int nt0  = blockIdx.x * GBN;

    // ---- staging geometry: thread owns 4 A-chunks + 2 W-chunks (16B each) ----
    const int srow  = tid >> 2;          // 0..31
    const int sch   = tid & 3;           // 16B chunk within a 64B row-slice
    const int voffA = srow * lda + sch * 8;   // per-thread elem offset within tile rows
    const int voffW = srow * ldw + sch * 8;
    // LDS byte addresses (generic-pointer low 32 bits == LDS offset)
    const unsigned ldsA = (unsigned)(uintptr_t)&sA[0][srow * LSTR + sch * 8];
    const unsigned ldsW = (unsigned)(uintptr_t)&sW[0][srow * LSTR + sch * 8];
    constexpr unsigned BUFA   = sizeof(sA[0]);       // bytes per A buffer
    constexpr unsigned BUFW   = sizeof(sW[0]);       // bytes per W buffer
    constexpr unsigned ROW32B = 32u * LSTR * 2u;     // 32 LDS rows in bytes

    // async-DMA one 32-deep K-slice into LDS buffer `buf`
    auto stage_async = [&](int ks, int buf) {
        // uniform (scalar) base select per K-slice: whole slice on one side of splitK
        const u16* ab = (ks < splitK) ? A0 + (size_t)mt0 * lda + ks
                                      : A1 + (size_t)mt0 * lda + (ks - splitK);
        const u16* wb = W + (size_t)nt0 * ldw + ks;
        const unsigned la = ldsA + (unsigned)buf * BUFA;
        const unsigned lw = ldsW + (unsigned)buf * BUFW;
        #pragma unroll
        for (int c = 0; c < 4; ++c)
            asm volatile("global_load_async_to_lds_b128 %0, %1, off"
                         :: "v"(la + c * ROW32B), "v"(ab + voffA + c * 32 * lda)
                         : "memory");
        #pragma unroll
        for (int c = 0; c < 2; ++c)
            asm volatile("global_load_async_to_lds_b128 %0, %1, off"
                         :: "v"(lw + c * ROW32B), "v"(wb + voffW + c * 32 * ldw)
                         : "memory");
    };

    v8f acc[2][4];
    #pragma unroll
    for (int i = 0; i < 2; ++i)
        #pragma unroll
        for (int j = 0; j < 4; ++j)
            #pragma unroll
            for (int v = 0; v < 8; ++v) acc[i][j][v] = 0.0f;

    stage_async(0, 0);
    wait_async0();
    __syncthreads();

    const int hf = lane >> 4;            // lane half selects K sub-block
    const int lr = lane & 15;            // row (A) / col (B) within 16

    for (int ks = 0, buf = 0; ks < K; ks += GBK, buf ^= 1) {
        const bool more = (ks + GBK) < K;
        if (more) stage_async(ks + GBK, buf ^ 1);        // DMA overlaps WMMAs below

        union { v16bf v; uint4 q[2]; } af[2], bw[4];
        #pragma unroll
        for (int i = 0; i < 2; ++i) {                    // A 16x32: q0 K=hf*8.., q1 K=16+hf*8..
            const u16* p = &sA[buf][(wave * 32 + i * 16 + lr) * LSTR + hf * 8];
            af[i].q[0] = *(const uint4*)p;
            af[i].q[1] = *(const uint4*)(p + 16);
        }
        #pragma unroll
        for (int j = 0; j < 4; ++j) {                    // B 32x16: lane=col n -> W row n
            const u16* p = &sW[buf][(j * 16 + lr) * LSTR + hf * 8];
            bw[j].q[0] = *(const uint4*)p;
            bw[j].q[1] = *(const uint4*)(p + 16);
        }
        #pragma unroll
        for (int i = 0; i < 2; ++i)
            #pragma unroll
            for (int j = 0; j < 4; ++j)
                acc[i][j] = __builtin_amdgcn_wmma_f32_16x16x32_bf16(
                    false, af[i].v, false, bw[j].v, (short)0, acc[i][j], false, false);

        if (more) wait_async0();                         // next buffer resident
        __syncthreads();
    }

    #pragma unroll
    for (int i = 0; i < 2; ++i)
        #pragma unroll
        for (int j = 0; j < 4; ++j) {
            const int n = nt0 + j * 16 + lr;
            const float bv = bias ? bias[n] : 0.0f;
            #pragma unroll
            for (int v = 0; v < 8; ++v) {               // D layout: m = v + 8*(lane/16)
                const int m = mt0 + wave * 32 + i * 16 + hf * 8 + v;
                float val = acc[i][j][v] + bv;
                if (flags & 1) val = fmaxf(val, 0.0f);
                if (flags & 2) ((float*)outp)[(size_t)m * ldo + n] = val;
                else           ((u16*)outp)[(size_t)m * ldo + n]  = f2bf(val);
            }
        }
}

// ---------------------------------------------------------------------------
// Pointwise / reduction kernels
// ---------------------------------------------------------------------------
__global__ void zkld_kernel(const float* __restrict__ eh, const float* __restrict__ ph,
                            const float* __restrict__ eps_t, u16* __restrict__ zbf,
                            float* __restrict__ kpart)
{
    __shared__ float red[256];
    const int i = blockIdx.x * 256 + threadIdx.x;   // 64 blocks -> 256*64 pairs
    const int b = i >> 6, d = i & 63;
    const float em  = eh[b * 128 + d];
    const float elv = fmaxf(eh[b * 128 + 64 + d], 0.0f);
    const float pm  = ph[b * 128 + d];
    const float plv = fmaxf(ph[b * 128 + 64 + d], 0.0f);
    const float e   = eps_t[b * 64 + d];
    zbf[b * 64 + d] = f2bf(em + expf(0.5f * elv) * e);
    const float dm = em - pm;
    red[threadIdx.x] = 0.5f * (plv - elv - 1.0f + (expf(elv) + dm * dm) * expf(-plv));
    __syncthreads();
    for (int s = 128; s > 0; s >>= 1) {
        if (threadIdx.x < s) red[threadIdx.x] += red[threadIdx.x + s];
        __syncthreads();
    }
    if (threadIdx.x == 0) kpart[blockIdx.x] = red[0];
}

__global__ void gmm_kernel(const float* __restrict__ dh, const float* __restrict__ y,
                           int t, float* __restrict__ llpart)
{
    __shared__ float red[64];
    const int b = blockIdx.x, tid = threadIdx.x;
    const float* row = dh + (size_t)b * 128;
    const float logit = (tid < 40) ? row[80 + tid] : -1e30f;
    red[tid] = logit; __syncthreads();
    for (int s = 32; s > 0; s >>= 1) { if (tid < s) red[tid] = fmaxf(red[tid], red[tid + s]); __syncthreads(); }
    const float mx = red[0]; __syncthreads();
    const float ex = (tid < 40) ? expf(logit - mx) : 0.0f;
    red[tid] = ex; __syncthreads();
    for (int s = 32; s > 0; s >>= 1) { if (tid < s) red[tid] += red[tid + s]; __syncthreads(); }
    const float denom = red[0]; __syncthreads();
    float contrib = 0.0f;
    if (tid < 40) {
        const float pi = ex / denom;
        const float mu = row[tid];
        const float lv = fmaxf(row[40 + tid], 0.0f);
        const int   d  = tid / 5;                      // col = d*5 + k
        const float yv = y[(size_t)b * 8 * TD + d * TD + t];
        const float df = yv - mu;
        contrib = pi * (-0.5f * (1.8378770664093453f + lv + df * df * expf(-lv)));
    }
    red[tid] = contrib; __syncthreads();
    for (int s = 32; s > 0; s >>= 1) { if (tid < s) red[tid] += red[tid + s]; __syncthreads(); }
    if (tid == 0) llpart[b] = red[0];
}

__global__ void gru_pw_kernel(const float* __restrict__ gx, const float* __restrict__ gh,
                              float* __restrict__ hstate, u16* __restrict__ hbf)
{
    const int i = blockIdx.x * 256 + threadIdx.x;   // 512 blocks -> 256*512
    const int b = i >> 9, j = i & 511;
    const size_t base = (size_t)b * 1536;
    float r  = gx[base + j]        + gh[base + j];
    float zg = gx[base + 512 + j]  + gh[base + 512 + j];
    const float nx = gx[base + 1024 + j];
    const float nh = gh[base + 1024 + j];
    r  = 1.0f / (1.0f + expf(-r));
    zg = 1.0f / (1.0f + expf(-zg));
    const float n  = tanhf(nx + r * nh);
    const float hn = (1.0f - zg) * n + zg * hstate[i];
    hstate[i] = hn;
    hbf[i]    = f2bf(hn);
}

__global__ void cvt_pad_kernel(const float* __restrict__ src, u16* __restrict__ dst,
                               int R, int C, int Cp)
{
    const int i = blockIdx.x * 256 + threadIdx.x;
    if (i >= R * Cp) return;
    const int r = i / Cp, c = i % Cp;
    dst[i] = f2bf(c < C ? src[(size_t)r * C + c] : 0.0f);
}

__global__ void pack_heads_kernel(const float* w0, const float* b0, int n0,
                                  const float* w1, const float* b1, int n1,
                                  const float* w2, const float* b2, int n2,
                                  u16* __restrict__ Wd, float* __restrict__ Bd)
{
    const int i = blockIdx.x * 256 + threadIdx.x;   // 128*512
    if (i >= 128 * 512) return;
    const int r = i >> 9, c = i & 511;
    float v = 0.0f; const float* bs = nullptr; int br = 0;
    if (r < n0)                      { v = w0[(size_t)r * 512 + c];              bs = b0; br = r; }
    else if (r < n0 + n1)            { v = w1[(size_t)(r - n0) * 512 + c];       bs = b1; br = r - n0; }
    else if (w2 && r < n0 + n1 + n2) { v = w2[(size_t)(r - n0 - n1) * 512 + c];  bs = b2; br = r - n0 - n1; }
    Wd[i] = f2bf(v);
    if (c == 0) Bd[r] = bs ? bs[br] : 0.0f;
}

__global__ void cvt_seq_kernel(const float* __restrict__ x, u16* __restrict__ xb, int D)
{
    const int i = blockIdx.x * 256 + threadIdx.x;   // T*B*32
    const int t = i >> 13;
    const int rem = i & 8191;
    const int b = rem >> 5, c = rem & 31;
    xb[i] = f2bf(c < D ? x[(size_t)b * D * TD + c * TD + t] : 0.0f);
}

__global__ void init_state_kernel(float* h0f, float* h1f, u16* h0b, u16* h1b)
{
    const int i = blockIdx.x * 256 + threadIdx.x;   // 512 blocks -> 256*512
    h0f[i] = 0.0f; h1f[i] = 0.0f; h0b[i] = 0; h1b[i] = 0;
}

__global__ void reduce_loss_kernel(const float* __restrict__ kp, int nk,
                                   const float* __restrict__ lp, int nl,
                                   float* __restrict__ out)
{
    __shared__ float red[256];
    float s = 0.0f;
    for (int i = threadIdx.x; i < nk; i += 256) s += kp[i];
    for (int i = threadIdx.x; i < nl; i += 256) s -= lp[i];
    red[threadIdx.x] = s; __syncthreads();
    for (int st = 128; st > 0; st >>= 1) {
        if (threadIdx.x < st) red[threadIdx.x] += red[threadIdx.x + st];
        __syncthreads();
    }
    if (threadIdx.x == 0) out[0] = red[0];
}

// ---------------------------------------------------------------------------
// Host schedule
// ---------------------------------------------------------------------------
extern "C" void kernel_launch(void* const* d_in, const int* in_sizes, int n_in,
                              void* d_out, int out_size, void* d_ws, size_t ws_size,
                              hipStream_t stream)
{
    const float* u   = (const float*)d_in[0];
    const float* y   = (const float*)d_in[1];
    const float* eps = (const float*)d_in[2];
    auto P = [&](int i) { return (const float*)d_in[i]; };

    char* cur = (char*)d_ws;
    auto alloc = [&](size_t bytes) -> char* {
        char* p = cur; cur += (bytes + 255) & ~size_t(255); return p;
    };
    auto au16 = [&](size_t n) { return (u16*)alloc(n * 2); };
    auto af32 = [&](size_t n) { return (float*)alloc(n * 4); };

    // bf16 weights (K padded to mult of 32; heads packed to 128 rows)
    u16 *w_py1 = au16(512*32),   *w_py2 = au16(512*512);
    u16 *w_pu1 = au16(512*32),   *w_pu2 = au16(512*512);
    u16 *w_pz1 = au16(512*64),   *w_pz2 = au16(512*512);
    u16 *w_e1  = au16(512*1024), *w_e2  = au16(512*512);
    u16 *w_d1  = au16(512*1024), *w_d2  = au16(512*512);
    u16 *w_pr1 = au16(512*512),  *w_pr2 = au16(512*512);
    u16 *w_eh  = au16(128*512);  float* b_eh = af32(128);
    u16 *w_ph  = au16(128*512);  float* b_ph = af32(128);
    u16 *w_dh  = au16(128*512);  float* b_dh = af32(128);
    u16 *w_gx0 = au16(1536*1024), *w_gh0 = au16(1536*512);
    u16 *w_gx1 = au16(1536*512),  *w_gh1 = au16(1536*512);
    // activations
    u16 *ybf = au16((size_t)TD*BD*32), *ubf = au16((size_t)TD*BD*32);
    u16 *tmp = au16(BD*HD);
    u16 *phiy = au16(BD*HD), *phiu = au16(BD*HD), *phiz = au16(BD*HD);
    u16 *enct = au16(BD*HD), *priort = au16(BD*HD), *dect = au16(BD*HD);
    u16 *h0b = au16(BD*HD), *h1b = au16(BD*HD), *zbf = au16(BD*ZD);
    float *h0f = af32(BD*HD), *h1f = af32(BD*HD);
    float *ehe = af32(BD*128), *phe = af32(BD*128), *dhe = af32(BD*128);
    float *gxf = af32((size_t)BD*1536), *ghf = af32((size_t)BD*1536);
    float *kpart = af32((size_t)TD*64), *lpart = af32((size_t)TD*BD);

    auto cvt = [&](const float* s, u16* d, int R, int C, int Cp) {
        const int n = R * Cp;
        cvt_pad_kernel<<<(n + 255) / 256, 256, 0, stream>>>(s, d, R, C, Cp);
    };
    cvt(P(3),  w_py1, 512, 8, 32);     cvt(P(5),  w_py2, 512, 512, 512);
    cvt(P(7),  w_pu1, 512, 8, 32);     cvt(P(9),  w_pu2, 512, 512, 512);
    cvt(P(11), w_pz1, 512, 64, 64);    cvt(P(13), w_pz2, 512, 512, 512);
    cvt(P(15), w_e1, 512, 1024, 1024); cvt(P(17), w_e2, 512, 512, 512);
    cvt(P(19), w_d1, 512, 1024, 1024); cvt(P(21), w_d2, 512, 512, 512);
    cvt(P(23), w_pr1, 512, 512, 512);  cvt(P(25), w_pr2, 512, 512, 512);
    cvt(P(41), w_gx0, 1536, 1024, 1024);
    cvt(P(42), w_gh0, 1536, 512, 512);
    cvt(P(43), w_gx1, 1536, 512, 512);
    cvt(P(44), w_gh1, 1536, 512, 512);
    pack_heads_kernel<<<256, 256, 0, stream>>>(P(27), P(28), 64, P(29), P(30), 64,
                                               nullptr, nullptr, 0, w_eh, b_eh);
    pack_heads_kernel<<<256, 256, 0, stream>>>(P(31), P(32), 64, P(33), P(34), 64,
                                               nullptr, nullptr, 0, w_ph, b_ph);
    pack_heads_kernel<<<256, 256, 0, stream>>>(P(35), P(36), 40, P(37), P(38), 40,
                                               P(39), P(40), 40, w_dh, b_dh);
    cvt_seq_kernel<<<8192, 256, 0, stream>>>(y, ybf, 8);
    cvt_seq_kernel<<<8192, 256, 0, stream>>>(u, ubf, 8);
    init_state_kernel<<<512, 256, 0, stream>>>(h0f, h1f, h0b, h1b);

    const int NOSPLIT = 1 << 30;
    auto gemm = [&](const u16* A0, const u16* A1, int lda, int splitK,
                    const u16* W, int ldw, const float* bias, void* out, int ldo,
                    int N, int K, int flags) {
        dim3 g(N / GBN, BD / GBM);
        gemm_bf16_wmma<<<g, 128, 0, stream>>>(A0, A1, lda, splitK,
                                              W, ldw, bias, out, ldo, N, K, flags);
    };

    for (int t = 0; t < TD; ++t) {
        const u16* yt = ybf + (size_t)t * BD * 32;
        const u16* ut = ubf + (size_t)t * BD * 32;
        // phi_y / phi_u  (Linear,ReLU,Linear)
        gemm(yt, nullptr, 32, NOSPLIT, w_py1, 32, P(4), tmp, 512, 512, 32, 1);
        gemm(tmp, nullptr, 512, NOSPLIT, w_py2, 512, P(6), phiy, 512, 512, 512, 0);
        gemm(ut, nullptr, 32, NOSPLIT, w_pu1, 32, P(8), tmp, 512, 512, 32, 1);
        gemm(tmp, nullptr, 512, NOSPLIT, w_pu2, 512, P(10), phiu, 512, 512, 512, 0);
        // enc([phi_y, h1]) + heads
        gemm(phiy, h1b, 512, 512, w_e1, 1024, P(16), tmp, 512, 512, 1024, 1);
        gemm(tmp, nullptr, 512, NOSPLIT, w_e2, 512, P(18), enct, 512, 512, 512, 1);
        gemm(enct, nullptr, 512, NOSPLIT, w_eh, 512, b_eh, ehe, 128, 128, 512, 2);
        // prior(h1) + heads
        gemm(h1b, nullptr, 512, NOSPLIT, w_pr1, 512, P(24), tmp, 512, 512, 512, 1);
        gemm(tmp, nullptr, 512, NOSPLIT, w_pr2, 512, P(26), priort, 512, 512, 512, 1);
        gemm(priort, nullptr, 512, NOSPLIT, w_ph, 512, b_ph, phe, 128, 128, 512, 2);
        // z = mu + exp(lv/2)*eps ; KLD partials
        zkld_kernel<<<64, 256, 0, stream>>>(ehe, phe, eps + (size_t)t * BD * ZD,
                                            zbf, kpart + (size_t)t * 64);
        // phi_z
        gemm(zbf, nullptr, 64, NOSPLIT, w_pz1, 64, P(12), tmp, 512, 512, 64, 1);
        gemm(tmp, nullptr, 512, NOSPLIT, w_pz2, 512, P(14), phiz, 512, 512, 512, 0);
        // dec([phi_z, h1]) + heads, GMM log-lik partials
        gemm(phiz, h1b, 512, 512, w_d1, 1024, P(20), tmp, 512, 512, 1024, 1);
        gemm(tmp, nullptr, 512, NOSPLIT, w_d2, 512, P(22), dect, 512, 512, 512, 1);
        gemm(dect, nullptr, 512, NOSPLIT, w_dh, 512, b_dh, dhe, 128, 128, 512, 2);
        gmm_kernel<<<256, 64, 0, stream>>>(dhe, y, t, lpart + (size_t)t * BD);
        // GRU layer 0 on [phi_u, phi_z]
        gemm(phiu, phiz, 512, 512, w_gx0, 1024, nullptr, gxf, 1536, 1536, 1024, 2);
        gemm(h0b, nullptr, 512, NOSPLIT, w_gh0, 512, nullptr, ghf, 1536, 1536, 512, 2);
        gru_pw_kernel<<<512, 256, 0, stream>>>(gxf, ghf, h0f, h0b);
        // GRU layer 1 on h0
        gemm(h0b, nullptr, 512, NOSPLIT, w_gx1, 512, nullptr, gxf, 1536, 1536, 512, 2);
        gemm(h1b, nullptr, 512, NOSPLIT, w_gh1, 512, nullptr, ghf, 1536, 1536, 512, 2);
        gru_pw_kernel<<<512, 256, 0, stream>>>(gxf, ghf, h1f, h1b);
    }
    reduce_loss_kernel<<<1, 256, 0, stream>>>(kpart, TD * 64, lpart, TD * BD,
                                              (float*)d_out);
}
